// NemotronHMamba2Mixer_83219286328136
// MI455X (gfx1250) — matrix operand order repaired and checked by reading, not compiled, not run on hardware
//
#include <hip/hip_runtime.h>

// ---------------------------------------------------------------------------
// NemotronH Mamba2 mixer for MI455X (gfx1250, wave32, WMMA).
// All matmuls (in_proj, out_proj, CB^T, L.CB^T @ X, Xd^T @ Bd, C @ prev^T)
// run on v_wmma_f32_16x16x32_bf16 with fp32 accumulators.
// GEMM: 512 threads / 16 waves, wave tile 64x32 -> low register pressure
// (no scratch spills), fragment-swizzled LDS (pure ds_load_b128 fragments),
// double buffering, global_prefetch_b8.
// ---------------------------------------------------------------------------

#define NB     2
#define NS     2048
#define NHID   4096
#define NH     128
#define NP     64
#define NN     128
#define NG     8
#define NKC    4              // conv kernel width
#define NINTER 8192           // NH*NP
#define NCONV  10240          // NINTER + 2*NG*NN
#define NPROJ  18560          // NINTER + NCONV + NH
#define NCHUNK 128
#define NCH    16             // NS / NCHUNK
#define GRP    1024           // NINTER / NG
#define EPSF   1e-5f

typedef __bf16 bf16_t;
typedef __attribute__((ext_vector_type(16))) __bf16 bf16x16;
typedef __attribute__((ext_vector_type(8)))  float  f32x8;

static __device__ inline f32x8 zero8() {
  f32x8 z;
  #pragma unroll
  for (int i = 0; i < 8; ++i) z[i] = 0.0f;
  return z;
}

static __device__ inline bf16_t f2bf(float f) {
  union { float f; unsigned int u; } a; a.f = f;
  unsigned int r = a.u + 0x7FFFu + ((a.u >> 16) & 1u);   // round-to-nearest-even
  union { unsigned short s; bf16_t b; } o; o.s = (unsigned short)(r >> 16);
  return o.b;
}

static __device__ inline unsigned int f2bf_pack(float lo, float hi) {
  union { unsigned short s; bf16_t b; } a, b2;
  a.b = f2bf(lo); b2.b = f2bf(hi);
  return (unsigned int)a.s | ((unsigned int)b2.s << 16);
}

static __device__ inline float siluf(float v) { return v / (1.0f + __expf(-v)); }

union FragU { bf16x16 v; unsigned int u[8]; uint4 q[2]; };

// A-matrix fragment (16x32 bf16) from LDS stored [row][ldk], k-contiguous.
// Contiguous-k layout (matches load_frag_b): lane L (r=L&15, h=L>>4); dword j holds K pair at
// k = h*16 + 2j of row m0+r.
static __device__ inline bf16x16 load_frag_a(const bf16_t* base, int ldk,
                                             int m0, int k0, int lane) {
  FragU f;
  int r = lane & 15, h = lane >> 4;
  const unsigned int* row = (const unsigned int*)(base + (m0 + r) * ldk + k0);
  #pragma unroll
  for (int j = 0; j < 8; ++j) {
    int k = h * 16 + j * 2;
    f.u[j] = row[k >> 1];
  }
  return f.v;
}

// B-matrix fragment (32x16 bf16) from LDS stored [N][ldk] (i.e. B^T, k-contig).
// Contiguous-k layout (matches load_frag_a): lane L (n=L&15, kh=L>>4); dword j holds K pair kh*16 + 2j.
static __device__ inline bf16x16 load_frag_b(const bf16_t* base, int ldk,
                                             int n0, int k0, int lane) {
  FragU f;
  int n = lane & 15, kh = lane >> 4;
  const unsigned int* row = (const unsigned int*)(base + (n0 + n) * ldk + k0 + kh * 16);
  #pragma unroll
  for (int j = 0; j < 8; ++j) f.u[j] = row[j];
  return f.v;
}

// Fragment-swizzled load: each lane's 8 dwords are contiguous -> 2x ds_load_b128.
static __device__ inline bf16x16 load_frag_sw(const uint4* tile, int lane) {
  FragU f;
  f.q[0] = tile[lane * 2 + 0];
  f.q[1] = tile[lane * 2 + 1];
  return f.v;
}

static __device__ inline f32x8 wmma_bf16(bf16x16 a, bf16x16 b, f32x8 c) {
  return __builtin_amdgcn_wmma_f32_16x16x32_bf16(false, a, false, b,
                                                 (short)0, c, false, false);
}

// ---------------------------------------------------------------------------
// K0: fp32 -> bf16 conversion (vectorized, grid-stride; n multiple of 4)
// ---------------------------------------------------------------------------
__global__ __launch_bounds__(256) void cvt_bf16_kernel(const float* __restrict__ src,
                                                       bf16_t* __restrict__ dst,
                                                       size_t n) {
  size_t i = (size_t)blockIdx.x * blockDim.x + threadIdx.x;
  size_t stride = (size_t)gridDim.x * blockDim.x;
  size_t n4 = n >> 2;
  const float4* s4 = (const float4*)src;
  uint2* d4 = (uint2*)dst;
  for (; i < n4; i += stride) {
    float4 v = s4[i];
    uint2 o;
    o.x = f2bf_pack(v.x, v.y);
    o.y = f2bf_pack(v.z, v.w);
    d4[i] = o;
  }
}

// ---------------------------------------------------------------------------
// K1/K8: GEMM  C[M][Nn] = A[M][K] * W[Nn][K]^T  (bf16 in, fp32 out)
// block tile 256x128, 512 threads = 16 waves (4x4), wave tile 64x32.
// LDS tiles are stored in WMMA-fragment order: element (r,k) with c = k>>3
// lives at [tile r>>4][lane (c&1)*16 + (r&15)][dword (c>>1)*4 + ((k>>1)&3)].
// A 16B global chunk (r, c) is exactly one 16B LDS store; a fragment is
// exactly two ds_load_b128. Double-buffered, one barrier per K-step.
// Per-wave registers: acc 64 + frags 48 + staging 12 -> no scratch spills.
// ---------------------------------------------------------------------------
__global__ __launch_bounds__(512, 1) void gemm_bf16_kernel(const bf16_t* __restrict__ A,
                                                           const bf16_t* __restrict__ W,
                                                           float* __restrict__ Cout,
                                                           int M, int Nn, int Kd) {
  __shared__ uint4 As_sw[2][16][32][2];   // 32 KB: 16 m-tiles
  __shared__ uint4 Ws_sw[2][8][32][2];    // 16 KB: 8 n-tiles
  const int tid = threadIdx.x, lane = tid & 31, w = tid >> 5;   // 16 waves
  const int wm = w >> 2, wn = w & 3;                 // wave grid 4 (M) x 4 (N)
  const int bm = blockIdx.y * 256, bn = blockIdx.x * 128;

  f32x8 acc[4][2];
  #pragma unroll
  for (int i = 0; i < 4; ++i)
    #pragma unroll
    for (int j = 0; j < 2; ++j) acc[i][j] = zero8();

  const int nk = Kd >> 5;
  uint4 ra[2], rb1;

  // Per-thread staging chunks: A has 256 rows x 4 chunks (2/thread),
  // W has 128 rows x 4 chunks (1/thread).
  // prologue: tile 0
  #pragma unroll
  for (int t = 0; t < 2; ++t) {
    int a = tid + t * 512, r = a >> 2, c = a & 3;
    ra[t] = *(const uint4*)&A[(size_t)(bm + r) * Kd + c * 8];
  }
  {
    int r = tid >> 2, c = tid & 3;
    rb1 = *(const uint4*)&W[(size_t)(bn + r) * Kd + c * 8];
  }
  #pragma unroll
  for (int t = 0; t < 2; ++t) {
    int a = tid + t * 512, r = a >> 2, c = a & 3;
    As_sw[0][r >> 4][(c & 1) * 16 + (r & 15)][c >> 1] = ra[t];
  }
  {
    int r = tid >> 2, c = tid & 3;
    Ws_sw[0][r >> 4][(c & 1) * 16 + (r & 15)][c >> 1] = rb1;
  }
  __syncthreads();

  for (int kt = 0; kt < nk; ++kt) {
    const int cur = kt & 1;
    const int kofs = (kt + 1) << 5;
    if (kt + 1 < nk) {
      // issue next tile's global loads before compute (latency overlap)
      #pragma unroll
      for (int t = 0; t < 2; ++t) {
        int a = tid + t * 512, r = a >> 2, c = a & 3;
        ra[t] = *(const uint4*)&A[(size_t)(bm + r) * Kd + kofs + c * 8];
      }
      {
        int r = tid >> 2, c = tid & 3;
        rb1 = *(const uint4*)&W[(size_t)(bn + r) * Kd + kofs + c * 8];
      }
      if (kt + 2 < nk)   // speculative prefetch (global_prefetch_b8)
        __builtin_prefetch(&A[(size_t)(bm + (tid >> 2)) * Kd + kofs + 32], 0, 0);
    }

    bf16x16 af[4], bfr[2];
    #pragma unroll
    for (int im = 0; im < 4; ++im)
      af[im] = load_frag_sw(&As_sw[cur][wm * 4 + im][0][0], lane);
    #pragma unroll
    for (int jn = 0; jn < 2; ++jn)
      bfr[jn] = load_frag_sw(&Ws_sw[cur][wn * 2 + jn][0][0], lane);
    #pragma unroll
    for (int im = 0; im < 4; ++im)
      #pragma unroll
      for (int jn = 0; jn < 2; ++jn)
        acc[im][jn] = wmma_bf16(af[im], bfr[jn], acc[im][jn]);

    if (kt + 1 < nk) {
      const int nxt = cur ^ 1;
      #pragma unroll
      for (int t = 0; t < 2; ++t) {
        int a = tid + t * 512, r = a >> 2, c = a & 3;
        As_sw[nxt][r >> 4][(c & 1) * 16 + (r & 15)][c >> 1] = ra[t];
      }
      {
        int r = tid >> 2, c = tid & 3;
        Ws_sw[nxt][r >> 4][(c & 1) * 16 + (r & 15)][c >> 1] = rb1;
      }
    }
    __syncthreads();
  }

  const int nn = lane & 15, hh = lane >> 4;
  #pragma unroll
  for (int im = 0; im < 4; ++im)
    #pragma unroll
    for (int jn = 0; jn < 2; ++jn)
      #pragma unroll
      for (int r = 0; r < 8; ++r) {
        int mrow = bm + wm * 64 + im * 16 + r + 8 * hh;
        int ncol = bn + wn * 32 + jn * 16 + nn;
        Cout[(size_t)mrow * Nn + ncol] = acc[im][jn][r];
      }
}

// ---------------------------------------------------------------------------
// K2: depthwise causal conv (K=4) + bias + SiLU over the hBC slice of proj
// ---------------------------------------------------------------------------
__global__ __launch_bounds__(256) void conv_silu_kernel(const float* __restrict__ proj,
                                                        const float* __restrict__ cw,
                                                        const float* __restrict__ cb,
                                                        float* __restrict__ hBC) {
  size_t i = (size_t)blockIdx.x * blockDim.x + threadIdx.x;
  if (i >= (size_t)NB * NS * NCONV) return;
  int c = (int)(i % NCONV);
  size_t bs = i / NCONV;
  int s = (int)(bs % NS);
  int b = (int)(bs / NS);
  float acc = cb[c];
  #pragma unroll
  for (int k = 0; k < NKC; ++k) {
    int sp = s - (NKC - 1) + k;
    if (sp >= 0)
      acc += cw[c * NKC + k] *
             proj[((size_t)b * NS + sp) * NPROJ + NINTER + c];
  }
  hBC[i] = siluf(acc);
}

// ---------------------------------------------------------------------------
// K3a: dt = clip(softplus(dt_raw + dt_bias), 0, 100)
// ---------------------------------------------------------------------------
__global__ __launch_bounds__(256) void dt_kernel(const float* __restrict__ proj,
                                                 const float* __restrict__ dt_bias,
                                                 float* __restrict__ dtv) {
  size_t i = (size_t)blockIdx.x * blockDim.x + threadIdx.x;
  if (i >= (size_t)NB * NS * NH) return;
  int h = (int)(i % NH);
  size_t bs = i / NH;
  float raw = proj[bs * NPROJ + NINTER + NCONV + h] + dt_bias[h];
  float dt = (raw > 20.0f) ? raw : log1pf(__expf(raw));
  dt = fminf(fmaxf(dt, 0.0f), 100.0f);
  dtv[i] = dt;
}

// ---------------------------------------------------------------------------
// K3b: per-(b,h,chunk) inclusive cumsum of dt*A  ->  Acum[b][h][z][l]
// ---------------------------------------------------------------------------
__global__ __launch_bounds__(256) void cumsum_kernel(const float* __restrict__ dtv,
                                                     const float* __restrict__ A_log,
                                                     float* __restrict__ Acum) {
  int i = blockIdx.x * blockDim.x + threadIdx.x;
  if (i >= NB * NH * NCH) return;
  int z = i % NCH;
  int h = (i / NCH) % NH;
  int b = i / (NCH * NH);
  float A = -__expf(A_log[h]);
  const float* d = dtv + ((size_t)b * NS + (size_t)z * NCHUNK) * NH + h;
  float* out = Acum + ((size_t)(b * NH + h) * NCH + z) * NCHUNK;
  float run = 0.0f;
  for (int l = 0; l < NCHUNK; ++l) {
    run += d[(size_t)l * NH] * A;
    out[l] = run;
  }
}

// ---------------------------------------------------------------------------
// K4: per-(b,z,h) chunk: CB^T -> L-mask -> Y_diag, and chunk states.
//   matmul1: Sc[l][s] = sum_n C[l][n]*B[s][n]         (128x128xK128)
//   matmul2: Ydiag[l][p] = sum_s L.Sc[l][s]*Xd[s][p]  (128x64xK128)
//   matmul3: states[p][n] = sum_s Xd[s][p]*Bd[s][n]   (64x128xK128)
// ---------------------------------------------------------------------------
__global__ __launch_bounds__(256) void chunk_kernel(const float* __restrict__ hBC,
                                                    const float* __restrict__ dtv,
                                                    const float* __restrict__ Acum,
                                                    float* __restrict__ Y,
                                                    float* __restrict__ states) {
  const int h = blockIdx.x, z = blockIdx.y, b = blockIdx.z;
  const int g = h / (NH / NG);
  const int tid = threadIdx.x, lane = tid & 31, w = tid >> 5;
  const int s0 = z * NCHUNK;

  const float* Cg  = hBC + ((size_t)b * NS + s0) * NCONV + NINTER + NG * NN + g * NN;
  const float* Bg  = hBC + ((size_t)b * NS + s0) * NCONV + NINTER + g * NN;
  const float* Xg  = hBC + ((size_t)b * NS + s0) * NCONV + h * NP;
  const float* dtg = dtv + ((size_t)b * NS + s0) * NH + h;
  const float* Ac  = Acum + ((size_t)(b * NH + h) * NCH + z) * NCHUNK;

  // LDS carve (62 KB total)
  __shared__ __align__(16) char smem[1024 + 34816 + 9216 + 18432];
  float*  AcumS = (float*)smem;                               // 512 B
  float*  dsc   = (float*)(smem + 512);                       // 512 B
  bf16_t* CbL   = (bf16_t*)(smem + 1024);                     // [128][68]
  bf16_t* BbL   = (bf16_t*)(smem + 1024 + 17408);             // [128][68]
  bf16_t* Sc    = (bf16_t*)(smem + 1024);                     // [128][132] (overlays CbL/BbL)
  bf16_t* XdTl  = (bf16_t*)(smem + 1024 + 34816);             // [64][72]
  bf16_t* BtLl  = (bf16_t*)(smem + 1024 + 34816 + 9216);      // [128][72]

  if (tid < NCHUNK) AcumS[tid] = Ac[tid];
  __syncthreads();
  float alast = AcumS[NCHUNK - 1];
  if (tid < NCHUNK) dsc[tid] = __expf(alast - AcumS[tid]);
  __syncthreads();

  // ---- matmul1: scores = C * B^T -------------------------------------------
  f32x8 acc1[8];
  #pragma unroll
  for (int i = 0; i < 8; ++i) acc1[i] = zero8();

  for (int ks = 0; ks < 2; ++ks) {
    for (int i = tid; i < 128 * 64; i += 256) {
      int r = i >> 6, c = i & 63;
      CbL[r * 68 + c] = f2bf(Cg[(size_t)r * NCONV + ks * 64 + c]);
      BbL[r * 68 + c] = f2bf(Bg[(size_t)r * NCONV + ks * 64 + c]);
    }
    __syncthreads();
    #pragma unroll
    for (int k0 = 0; k0 < 64; k0 += 32) {
      bf16x16 af = load_frag_a(CbL, 68, w * 16, k0, lane);
      #pragma unroll
      for (int st = 0; st < 8; ++st) {
        bf16x16 bfr = load_frag_b(BbL, 68, st * 16, k0, lane);
        acc1[st] = wmma_bf16(af, bfr, acc1[st]);
      }
    }
    __syncthreads();
  }

  // Apply L = tril(exp(Acum[l]-Acum[s])) on accumulator, store bf16 scores.
  {
    const int nn = lane & 15, hh = lane >> 4;
    #pragma unroll
    for (int st = 0; st < 8; ++st)
      #pragma unroll
      for (int r = 0; r < 8; ++r) {
        int l = w * 16 + r + 8 * hh;
        int s = st * 16 + nn;
        float v = acc1[st][r];
        v = (l >= s) ? v * __expf(AcumS[l] - AcumS[s]) : 0.0f;
        Sc[l * 132 + s] = f2bf(v);
      }
  }
  __syncthreads();

  // ---- matmul2 (Y_diag) and matmul3 (states), shared s-staging -------------
  f32x8 acc2[4], acc3[4];
  #pragma unroll
  for (int i = 0; i < 4; ++i) { acc2[i] = zero8(); acc3[i] = zero8(); }

  for (int ks = 0; ks < 2; ++ks) {
    for (int i = tid; i < 64 * 64; i += 256) {     // XdT[p][sl] = x*dt (transposed)
      int p = i >> 6, sl = i & 63;
      int s = ks * 64 + sl;
      XdTl[p * 72 + sl] = f2bf(Xg[(size_t)s * NCONV + p] * dtg[(size_t)s * NH]);
    }
    for (int i = tid; i < 128 * 64; i += 256) {    // BtL[n][sl] = B*decay (transposed)
      int n = i >> 6, sl = i & 63;
      int s = ks * 64 + sl;
      BtLl[n * 72 + sl] = f2bf(Bg[(size_t)s * NCONV + n] * dsc[s]);
    }
    __syncthreads();
    #pragma unroll
    for (int k0 = 0; k0 < 64; k0 += 32) {
      // Y_diag: A = Sc (l-tile w), Bop = XdT  -> D[l][p]
      bf16x16 af2 = load_frag_a(Sc, 132, w * 16, ks * 64 + k0, lane);
      #pragma unroll
      for (int pt = 0; pt < 4; ++pt) {
        bf16x16 bf2 = load_frag_b(XdTl, 72, pt * 16, k0, lane);
        acc2[pt] = wmma_bf16(af2, bf2, acc2[pt]);
      }
      // states: A = XdT (p-tiles), Bop = BtL (n-tile w) -> D[p][n]
      bf16x16 bf3 = load_frag_b(BtLl, 72, w * 16, k0, lane);
      #pragma unroll
      for (int pt = 0; pt < 4; ++pt) {
        bf16x16 af3 = load_frag_a(XdTl, 72, pt * 16, k0, lane);
        acc3[pt] = wmma_bf16(af3, bf3, acc3[pt]);
      }
    }
    __syncthreads();
  }

  const int nn = lane & 15, hh = lane >> 4;
  #pragma unroll
  for (int pt = 0; pt < 4; ++pt)
    #pragma unroll
    for (int r = 0; r < 8; ++r) {
      int l = w * 16 + r + 8 * hh;
      int p = pt * 16 + nn;
      Y[((size_t)b * NS + s0 + l) * NINTER + h * NP + p] = acc2[pt][r];
    }
  float* st_out = states + (((size_t)b * NCH + z) * NH + h) * NP * NN;
  #pragma unroll
  for (int pt = 0; pt < 4; ++pt)
    #pragma unroll
    for (int r = 0; r < 8; ++r) {
      int p = pt * 16 + r + 8 * hh;
      int n = w * 16 + nn;
      st_out[(size_t)p * NN + n] = acc3[pt][r];
    }
}

// ---------------------------------------------------------------------------
// K5: inter-chunk state recurrence: prev[z] = prev[z-1]*exp(Alast[z-1]) + states[z-1]
// ---------------------------------------------------------------------------
__global__ __launch_bounds__(256) void recur_kernel(const float* __restrict__ states,
                                                    const float* __restrict__ Acum,
                                                    float* __restrict__ prevs) {
  size_t i = (size_t)blockIdx.x * blockDim.x + threadIdx.x;
  if (i >= (size_t)NB * NH * NP * NN) return;
  int n = (int)(i % NN);
  int p = (int)((i / NN) % NP);
  int h = (int)((i / ((size_t)NN * NP)) % NH);
  int b = (int)(i / ((size_t)NN * NP * NH));
  float run = 0.0f;
  for (int z = 0; z < NCH; ++z) {
    size_t si = ((((size_t)b * NCH + z) * NH + h) * NP + p) * NN + n;
    prevs[si] = run;
    float alast = Acum[((size_t)(b * NH + h) * NCH + z) * NCHUNK + NCHUNK - 1];
    run = run * __expf(alast) + states[si];
  }
}

// ---------------------------------------------------------------------------
// K6: Y += exp(Acum[l]) * (C @ prev^T) + x*D   per (b,z,h)
// ---------------------------------------------------------------------------
__global__ __launch_bounds__(256) void yoff_kernel(const float* __restrict__ hBC,
                                                   const float* __restrict__ Acum,
                                                   const float* __restrict__ prevs,
                                                   const float* __restrict__ Dv,
                                                   float* __restrict__ Y) {
  const int h = blockIdx.x, z = blockIdx.y, b = blockIdx.z;
  const int g = h / (NH / NG);
  const int tid = threadIdx.x, lane = tid & 31, w = tid >> 5;
  const int s0 = z * NCHUNK;

  const float* Cg = hBC + ((size_t)b * NS + s0) * NCONV + NINTER + NG * NN + g * NN;
  const float* Xg = hBC + ((size_t)b * NS + s0) * NCONV + h * NP;
  const float* Ac = Acum + ((size_t)(b * NH + h) * NCH + z) * NCHUNK;
  const float* Pv = prevs + (((size_t)b * NCH + z) * NH + h) * NP * NN;

  __shared__ __align__(16) char sm[512 + 17408 + 8704];
  float*  AcumS = (float*)sm;                       // 512 B
  bf16_t* CbL   = (bf16_t*)(sm + 512);              // [128][68]
  bf16_t* PbL   = (bf16_t*)(sm + 512 + 17408);      // [64][68]

  if (tid < NCHUNK) AcumS[tid] = Ac[tid];

  f32x8 acc[4];
  #pragma unroll
  for (int i = 0; i < 4; ++i) acc[i] = zero8();

  for (int ks = 0; ks < 2; ++ks) {
    for (int i = tid; i < 128 * 64; i += 256) {
      int r = i >> 6, c = i & 63;
      CbL[r * 68 + c] = f2bf(Cg[(size_t)r * NCONV + ks * 64 + c]);
    }
    for (int i = tid; i < 64 * 64; i += 256) {
      int p = i >> 6, c = i & 63;
      PbL[p * 68 + c] = f2bf(Pv[(size_t)p * NN + ks * 64 + c]);
    }
    __syncthreads();
    #pragma unroll
    for (int k0 = 0; k0 < 64; k0 += 32) {
      bf16x16 af = load_frag_a(CbL, 68, w * 16, k0, lane);
      #pragma unroll
      for (int pt = 0; pt < 4; ++pt) {
        bf16x16 bfr = load_frag_b(PbL, 68, pt * 16, k0, lane);
        acc[pt] = wmma_bf16(af, bfr, acc[pt]);
      }
    }
    __syncthreads();
  }

  const float Dh = Dv[h];
  const int nn = lane & 15, hh = lane >> 4;
  #pragma unroll
  for (int pt = 0; pt < 4; ++pt)
    #pragma unroll
    for (int r = 0; r < 8; ++r) {
      int l = w * 16 + r + 8 * hh;
      int p = pt * 16 + nn;
      size_t yi = ((size_t)b * NS + s0 + l) * NINTER + h * NP + p;
      float xs = Xg[(size_t)l * NCONV + p];
      Y[yi] = Y[yi] + __expf(AcumS[l]) * acc[pt][r] + xs * Dh;
    }
}

// ---------------------------------------------------------------------------
// K7: g = Y * silu(gate); grouped RMS norm (groups of 1024) * norm_w -> bf16
// ---------------------------------------------------------------------------
__global__ __launch_bounds__(256) void gatenorm_kernel(const float* __restrict__ Y,
                                                       const float* __restrict__ proj,
                                                       const float* __restrict__ norm_w,
                                                       bf16_t* __restrict__ outb) {
  const int gblk = blockIdx.x, s = blockIdx.y, b = blockIdx.z;
  const int tid = threadIdx.x;
  const size_t base  = ((size_t)b * NS + s) * NINTER + (size_t)gblk * GRP;
  const size_t pbase = ((size_t)b * NS + s) * NPROJ + (size_t)gblk * GRP;

  float vals[4];
  float ss = 0.0f;
  #pragma unroll
  for (int i = 0; i < 4; ++i) {
    int idx = tid + i * 256;
    float gt = proj[pbase + idx];
    float v = Y[base + idx] * siluf(gt);
    vals[i] = v;
    ss += v * v;
  }
  __shared__ float red[256];
  red[tid] = ss;
  __syncthreads();
  #pragma unroll
  for (int st = 128; st > 0; st >>= 1) {
    if (tid < st) red[tid] += red[tid + st];
    __syncthreads();
  }
  float scale = rsqrtf(red[0] / (float)GRP + EPSF);
  #pragma unroll
  for (int i = 0; i < 4; ++i) {
    int idx = tid + i * 256;
    outb[base + idx] = f2bf(vals[i] * scale * norm_w[gblk * GRP + idx]);
  }
}

// ---------------------------------------------------------------------------
// Host launch
// ---------------------------------------------------------------------------
static inline size_t alignup(size_t x) { return (x + 255) & ~(size_t)255; }

extern "C" void kernel_launch(void* const* d_in, const int* in_sizes, int n_in,
                              void* d_out, int out_size, void* d_ws, size_t ws_size,
                              hipStream_t stream) {
  const float* x          = (const float*)d_in[0];
  const float* in_proj_w  = (const float*)d_in[1];
  const float* conv_w     = (const float*)d_in[2];
  const float* conv_b     = (const float*)d_in[3];
  const float* dt_bias    = (const float*)d_in[4];
  const float* A_log      = (const float*)d_in[5];
  const float* Dv         = (const float*)d_in[6];
  const float* norm_w     = (const float*)d_in[7];
  const float* out_proj_w = (const float*)d_in[8];
  float* out = (float*)d_out;

  char* ws = (char*)d_ws;
  size_t off = 0;
  auto carve = [&](size_t bytes) { char* p = ws + off; off = alignup(off + bytes); return p; };

  bf16_t* xb     = (bf16_t*)carve((size_t)NB * NS * NHID * 2);
  bf16_t* winb   = (bf16_t*)carve((size_t)NPROJ * NHID * 2);
  bf16_t* woutb  = (bf16_t*)carve((size_t)NHID * NINTER * 2);
  float*  proj   = (float*) carve((size_t)NB * NS * NPROJ * 4);
  float*  hBC    = (float*) carve((size_t)NB * NS * NCONV * 4);
  float*  dtv    = (float*) carve((size_t)NB * NS * NH * 4);
  float*  Acum   = (float*) carve((size_t)NB * NH * NCH * NCHUNK * 4);
  float*  states = (float*) carve((size_t)NB * NCH * NH * NP * NN * 4);
  float*  prevs  = (float*) carve((size_t)NB * NCH * NH * NP * NN * 4);
  float*  Ybuf   = (float*) carve((size_t)NB * NS * NINTER * 4);
  bf16_t* ygb    = (bf16_t*)carve((size_t)NB * NS * NINTER * 2);
  (void)ws_size; (void)in_sizes; (void)n_in; (void)out_size;

  // K0: bf16 conversions
  cvt_bf16_kernel<<<2048, 256, 0, stream>>>(x, xb, (size_t)NB * NS * NHID);
  cvt_bf16_kernel<<<4096, 256, 0, stream>>>(in_proj_w, winb, (size_t)NPROJ * NHID);
  cvt_bf16_kernel<<<4096, 256, 0, stream>>>(out_proj_w, woutb, (size_t)NHID * NINTER);

  // K1: in_proj GEMM [4096 x 18560 x 4096]
  gemm_bf16_kernel<<<dim3(NPROJ / 128, (NB * NS) / 256), 512, 0, stream>>>(
      xb, winb, proj, NB * NS, NPROJ, NHID);

  // K2: depthwise conv + SiLU
  conv_silu_kernel<<<(unsigned)(((size_t)NB * NS * NCONV) / 256), 256, 0, stream>>>(
      proj, conv_w, conv_b, hBC);

  // K3: dt + cumsum
  dt_kernel<<<(unsigned)(((size_t)NB * NS * NH) / 256), 256, 0, stream>>>(proj, dt_bias, dtv);
  cumsum_kernel<<<(NB * NH * NCH + 255) / 256, 256, 0, stream>>>(dtv, A_log, Acum);

  // K4: per-chunk WMMA (Y_diag + states)
  chunk_kernel<<<dim3(NH, NCH, NB), 256, 0, stream>>>(hBC, dtv, Acum, Ybuf, states);

  // K5: inter-chunk recurrence
  recur_kernel<<<(unsigned)(((size_t)NB * NH * NP * NN) / 256), 256, 0, stream>>>(
      states, Acum, prevs);

  // K6: Y_off + skip (x*D)
  yoff_kernel<<<dim3(NH, NCH, NB), 256, 0, stream>>>(hBC, Acum, prevs, Dv, Ybuf);

  // K7: gate + group RMS norm -> bf16
  gatenorm_kernel<<<dim3(NG, NS, NB), 256, 0, stream>>>(Ybuf, proj, norm_w, ygb);

  // K8: out_proj GEMM [4096 x 4096 x 8192]
  gemm_bf16_kernel<<<dim3(NHID / 128, (NB * NS) / 256), 512, 0, stream>>>(
      ygb, woutb, out, NB * NS, NHID, NINTER);
}
